// JKNet_44478681317638
// MI455X (gfx1250) — matrix operand order repaired and checked by reading, not compiled, not run on hardware
//
#include <hip/hip_runtime.h>

typedef __attribute__((ext_vector_type(16))) __bf16          v16bf;
typedef __attribute__((ext_vector_type(8)))  float           v8f;
typedef __attribute__((ext_vector_type(8)))  unsigned short  ushort8;

#define N_NODES 50000
#define N_EDGES 1600000LL
#define FEATS   512
#define HID     256
#define NLAYERS 4
#define CLASSES 64
#define CATW    (NLAYERS * HID)   /* 1024 */

#define BT_STRIDE 40              /* padded [n][k] stride: 80B rows, 16B-aligned half-rows */

union Frag16 { v16bf v; unsigned int u[8]; };

__device__ __forceinline__ unsigned short f32_to_bf16(float f) {
  unsigned int u = __float_as_uint(f);
  unsigned int r = 0x7FFFu + ((u >> 16) & 1u);  // round-to-nearest-even
  return (unsigned short)((u + r) >> 16);
}
__device__ __forceinline__ float bf16_to_f32(unsigned short h) {
  return __uint_as_float(((unsigned int)h) << 16);
}

// ---------------- elementwise f32 -> bf16 ----------------
__global__ __launch_bounds__(256) void k_cvt_bf16(const float* __restrict__ in,
                                                  unsigned short* __restrict__ out,
                                                  long long n) {
  long long i = (long long)blockIdx.x * blockDim.x + threadIdx.x;
  long long stride = (long long)gridDim.x * blockDim.x;
  for (; i < n; i += stride) out[i] = f32_to_bf16(in[i]);
}

// ---------------- degree / dinv ----------------
__global__ __launch_bounds__(256) void k_deg_init(float* __restrict__ deg, int n) {
  int i = blockIdx.x * blockDim.x + threadIdx.x;
  if (i < n) deg[i] = 1.0f;            // self-loop contribution
}
__global__ __launch_bounds__(256) void k_deg_edges(const long long* __restrict__ dst,
                                                   long long nE, float* __restrict__ deg) {
  long long e = (long long)blockIdx.x * blockDim.x + threadIdx.x;
  if (e < nE) atomicAdd(&deg[(int)dst[e]], 1.0f);
}
__global__ __launch_bounds__(256) void k_deg_finalize(float* __restrict__ deg, int n) {
  int i = blockIdx.x * blockDim.x + threadIdx.x;
  if (i < n) { float d = deg[i]; deg[i] = (d > 0.0f) ? rsqrtf(d) : 0.0f; }
}

// ---------------- WMMA GEMM: C[M,BN] = A[M,K](bf16) x B[K,BN](bf16) ----------------
// Block = 128 threads (4 waves). Block tile: 16 (M) x BN (N), BN == exact N.
// Wave tile: 16 x (BN/4) => BN/64 accumulators per wave sharing one A fragment.
// All shapes compile-time => no guards, no divergence, EXEC all-1s at every WMMA.
// MODE 0: store bf16 to outb.  MODE 1: store f32 (acc + bias[n]) to outf.
template <int MODE, int BN>
__global__ __launch_bounds__(128) void k_gemm_wmma(
    const unsigned short* __restrict__ A, int lda,
    const unsigned short* __restrict__ B, int K,
    unsigned short* __restrict__ outb, float* __restrict__ outf,
    const float* __restrict__ bias, int ldc) {
  constexpr int WCOLS  = BN / 4;     // columns per wave
  constexpr int WFRAGS = WCOLS / 16; // 16x16 fragments per wave

  __shared__ __align__(16) unsigned short As[16 * 32];           // [m][k] row-major
  __shared__ __align__(16) unsigned short Bts[BN * BT_STRIDE];   // [n][k] transposed

  const int m0   = blockIdx.x * 16;
  const int t    = threadIdx.x;
  const int wave = t >> 5;
  const int lane = t & 31;
  const int half = lane >> 4;
  const int lm   = lane & 15;

  v8f acc[WFRAGS] = {};

  for (int k0 = 0; k0 < K; k0 += 32) {
    // --- stage A tile (16x32): 4 bf16 (one u64) per thread ---
    {
      int e = t * 4;
      int r = e >> 5, c = e & 31;
      *(unsigned long long*)&As[r * 32 + c] =
          *(const unsigned long long*)&A[(size_t)(m0 + r) * lda + k0 + c];
    }
    // --- stage B tile (32 x BN) transposed into Bts[n][k] ---
    {
      int r  = t & 31;                 // k row 0..31
      int c0 = (t >> 5) * WCOLS;       // column group per thread-slice
      const unsigned short* Brow = &B[(size_t)(k0 + r) * BN + c0];
#pragma unroll
      for (int g = 0; g < WCOLS / 8; ++g) {
        ushort8 pv = *(const ushort8*)(Brow + g * 8);
#pragma unroll
        for (int i = 0; i < 8; ++i)
          Bts[(c0 + g * 8 + i) * BT_STRIDE + r] = pv[i];
      }
    }
    __syncthreads();

    // --- one A fragment, WFRAGS B fragments, WFRAGS independent WMMAs ---
    Frag16 a;
#pragma unroll
    for (int j = 0; j < 8; ++j) {
      // ISA bf16 A layout: lanes0-15 VGPR0-3 -> K=2j ; VGPR4-7 -> K=16+2(j-4); lanes16-31 +8
      int ka = 2 * (j & 3) + ((j >= 4) ? 16 : 0) + half * 8;
      a.u[j] = *(const unsigned int*)&As[lm * 32 + ka];
    }
#pragma unroll
    for (int f = 0; f < WFRAGS; ++f) {
      int nn = wave * WCOLS + f * 16;
      Frag16 b;
#pragma unroll
      for (int j = 0; j < 8; ++j) {
        // bf16 B layout matches A: VGPR0-3 -> K=2j ; VGPR4-7 -> K=16+2(j-4); lanes16-31 +8
        b.u[j] = *(const unsigned int*)&Bts[(nn + lm) * BT_STRIDE + 2 * (j & 3) + ((j >= 4) ? 16 : 0) + half * 8];
      }
      acc[f] = __builtin_amdgcn_wmma_f32_16x16x32_bf16(
          false, a.v, false, b.v, (short)0, acc[f], false, false);
    }
    __syncthreads();
  }

  // --- epilogue: C layout lane n = lane&15 ; VGPR j -> m = j + 8*(lane>>4) ---
#pragma unroll
  for (int f = 0; f < WFRAGS; ++f) {
    int n = wave * WCOLS + f * 16 + lm;
#pragma unroll
    for (int j = 0; j < 8; ++j) {
      int m = m0 + j + half * 8;
      if (MODE == 0) {
        outb[(size_t)m * ldc + n] = f32_to_bf16(acc[f][j]);
      } else {
        outf[(size_t)m * ldc + n] = acc[f][j] + bias[n];
      }
    }
  }
}

// ---------------- agg init with self-loop term: agg = t * dinv^2 ----------------
__global__ __launch_bounds__(256) void k_agg_self(const unsigned short* __restrict__ t,
                                                  const float* __restrict__ dinv,
                                                  float* __restrict__ agg) {
  int idx = blockIdx.x * blockDim.x + threadIdx.x;   // over N_NODES*HID = 12.8M
  int i = idx >> 8;
  float di = dinv[i];
  agg[idx] = bf16_to_f32(t[idx]) * di * di;
}

// ---------------- edge aggregation: one wave per edge, 8 feats/lane ----------------
__global__ __launch_bounds__(256) void k_edge_agg(const long long* __restrict__ src,
                                                  const long long* __restrict__ dst,
                                                  long long nE,
                                                  const unsigned short* __restrict__ t,
                                                  const float* __restrict__ dinv,
                                                  float* __restrict__ agg) {
  long long e = (long long)blockIdx.x * 8 + (threadIdx.x >> 5);
  if (e >= nE) return;
  int lane = threadIdx.x & 31;
  int s = (int)src[e], d = (int)dst[e];
  float norm = dinv[s] * dinv[d];
  const ushort8 v = *(const ushort8*)(t + (size_t)s * HID + lane * 8);
  float* arow = agg + (size_t)d * HID + lane * 8;
#pragma unroll
  for (int i = 0; i < 8; ++i)
    atomicAdd(&arow[i], bf16_to_f32(v[i]) * norm);
}

// ---------------- relu(agg + b) -> bf16 slice of cat buffer ----------------
__global__ __launch_bounds__(256) void k_bias_relu_cat(const float* __restrict__ agg,
                                                       const float* __restrict__ bias,
                                                       unsigned short* __restrict__ cat,
                                                       int layer) {
  int idx = blockIdx.x * blockDim.x + threadIdx.x;   // over N_NODES*HID
  int i = idx >> 8, f = idx & 255;
  float v = agg[idx] + bias[f];
  v = fmaxf(v, 0.0f);
  cat[(size_t)i * CATW + layer * HID + f] = f32_to_bf16(v);
}

// ---------------- log_softmax over 64 classes, one wave per row ----------------
__global__ __launch_bounds__(256) void k_log_softmax64(float* __restrict__ out, int n) {
  int row = blockIdx.x * 8 + (threadIdx.x >> 5);
  if (row >= n) return;
  int lane = threadIdx.x & 31;
  float* p = out + (size_t)row * CLASSES;
  float a = p[lane], b = p[lane + 32];
  float m = fmaxf(a, b);
  for (int o = 16; o > 0; o >>= 1) m = fmaxf(m, __shfl_xor(m, o, 32));
  float s = __expf(a - m) + __expf(b - m);
  for (int o = 16; o > 0; o >>= 1) s += __shfl_xor(s, o, 32);
  float ls = m + __logf(s);
  p[lane]      = a - ls;
  p[lane + 32] = b - ls;
}

// =====================================================================
extern "C" void kernel_launch(void* const* d_in, const int* in_sizes, int n_in,
                              void* d_out, int out_size, void* d_ws, size_t ws_size,
                              hipStream_t stream) {
  (void)in_sizes; (void)n_in; (void)out_size; (void)ws_size;
  const float*     x   = (const float*)d_in[0];
  const long long* ei  = (const long long*)d_in[1];     // int64 [2, E]
  const float*     W0  = (const float*)d_in[2];
  const float*     b0  = (const float*)d_in[3];
  const float*     Wh  = (const float*)d_in[4];
  const float*     bh  = (const float*)d_in[5];
  const float*     Wfc = (const float*)d_in[6];
  const float*     bfc = (const float*)d_in[7];
  float* out = (float*)d_out;

  const long long E = N_EDGES;
  const long long* srcp = ei;
  const long long* dstp = ei + E;

  // ---- carve workspace (256B aligned chunks) ----
  char* p = (char*)d_ws;
  auto carve = [&p](size_t bytes) -> void* {
    uintptr_t a = ((uintptr_t)p + 255) & ~(uintptr_t)255;
    p = (char*)(a + bytes);
    return (void*)a;
  };
  unsigned short* xbf  = (unsigned short*)carve((size_t)N_NODES * FEATS * 2);
  unsigned short* w0b  = (unsigned short*)carve((size_t)FEATS * HID * 2);
  unsigned short* whb  = (unsigned short*)carve((size_t)(NLAYERS - 1) * HID * HID * 2);
  unsigned short* wfcb = (unsigned short*)carve((size_t)CATW * CLASSES * 2);
  float*          dinv = (float*)carve((size_t)N_NODES * 4);
  unsigned short* t    = (unsigned short*)carve((size_t)N_NODES * HID * 2);
  float*          agg  = (float*)carve((size_t)N_NODES * HID * 4);
  unsigned short* cat  = (unsigned short*)carve((size_t)N_NODES * CATW * 2);

  // ---- precision conversion to bf16 ----
  k_cvt_bf16<<<4096, 256, 0, stream>>>(x,   xbf,  (long long)N_NODES * FEATS);
  k_cvt_bf16<<<512,  256, 0, stream>>>(W0,  w0b,  (long long)FEATS * HID);
  k_cvt_bf16<<<512,  256, 0, stream>>>(Wh,  whb,  (long long)(NLAYERS - 1) * HID * HID);
  k_cvt_bf16<<<256,  256, 0, stream>>>(Wfc, wfcb, (long long)CATW * CLASSES);

  // ---- symmetric normalization: dinv = rsqrt(indeg + 1) ----
  k_deg_init    <<<(N_NODES + 255) / 256, 256, 0, stream>>>(dinv, N_NODES);
  k_deg_edges   <<<(int)((E + 255) / 256), 256, 0, stream>>>(dstp, E, dinv);
  k_deg_finalize<<<(N_NODES + 255) / 256, 256, 0, stream>>>(dinv, N_NODES);

  // ---- GCN layers ----
  const int elemGrid = (N_NODES * HID) / 256;   // 50000
  dim3 gH(N_NODES / 16, 1);                     // block tile 16 x 256 == full N
  for (int l = 0; l < NLAYERS; ++l) {
    const unsigned short* Ain = (l == 0) ? xbf : (cat + (size_t)(l - 1) * HID);
    int lda = (l == 0) ? FEATS : CATW;
    const unsigned short* Bw = (l == 0) ? w0b : (whb + (size_t)(l - 1) * HID * HID);
    int K = (l == 0) ? FEATS : HID;
    k_gemm_wmma<0, HID><<<gH, 128, 0, stream>>>(Ain, lda, Bw, K,
                                                t, nullptr, nullptr, HID);
    k_agg_self <<<elemGrid, 256, 0, stream>>>(t, dinv, agg);
    k_edge_agg <<<(int)((E + 7) / 8), 256, 0, stream>>>(srcp, dstp, E, t, dinv, agg);
    const float* bias = (l == 0) ? b0 : (bh + (size_t)(l - 1) * HID);
    k_bias_relu_cat<<<elemGrid, 256, 0, stream>>>(agg, bias, cat, l);
  }

  // ---- classifier: cat[N,1024] @ Wfc[1024,64] + bfc -> d_out (f32) ----
  dim3 gF(N_NODES / 16, 1);
  k_gemm_wmma<1, CLASSES><<<gF, 128, 0, stream>>>(cat, CATW, wfcb, CATW,
                                                  nullptr, out, bfc, CLASSES);
  k_log_softmax64<<<(N_NODES + 7) / 8, 256, 0, stream>>>(out, N_NODES);
}
